// GRUEncoder_30872224924227
// MI455X (gfx1250) — matrix operand order repaired and checked by reading, not compiled, not run on hardware
//
#include <hip/hip_runtime.h>

// Sizes (fixed by the reference): B=128, T=512, D=512, H=512, L=256, 3H=1536, 2L=512
#define GB   128
#define GT   512
#define GD   512
#define GH   512
#define G3H  1536

typedef __attribute__((ext_vector_type(16))) __bf16 v16bf;
typedef __attribute__((ext_vector_type(8)))  float  v8f;

union BfFrag {
    unsigned short s[16];
    uint4          q[2];
    v16bf          v;
};

__device__ __forceinline__ unsigned short f2bf(float f) {
    unsigned u = __float_as_uint(f);
    unsigned r = u + 0x7FFFu + ((u >> 16) & 1u);   // round-to-nearest-even
    return (unsigned short)(r >> 16);
}

__device__ __forceinline__ v8f wmma_bf16(v16bf a, v16bf b, v8f c) {
    // D = A(16x32) x B(32x16) + C, f32 accumulate
    return __builtin_amdgcn_wmma_f32_16x16x32_bf16(
        /*neg_a=*/false, a, /*neg_b=*/false, b,
        /*c_mod=*/(short)0, c, /*reuse_a=*/false, /*reuse_b=*/false);
}

__device__ __forceinline__ float sigmoidf_fast(float x) {
    return 1.0f / (1.0f + __expf(-x));
}

// ---------------------------------------------------------------------------
// Kernel 0: convert weights fp32 -> bf16, zero h double-buffer + barrier ctr
// ---------------------------------------------------------------------------
__global__ void __launch_bounds__(256) gru_prep_kernel(
    const float* __restrict__ Wih, const float* __restrict__ Whh,
    const float* __restrict__ Wlin,
    unsigned short* __restrict__ wih_bf, unsigned short* __restrict__ whh_bf,
    unsigned short* __restrict__ wlin_bf,
    float* __restrict__ hbuf2, unsigned* __restrict__ ctr) {
    int i = blockIdx.x * 256 + threadIdx.x;
    if (i < G3H * GD) {                 // 786432
        wih_bf[i] = f2bf(Wih[i]);
        whh_bf[i] = f2bf(Whh[i]);
    }
    if (i < 2 * 256 * GH)               // Wlin: (2L, H) = 512*512
        wlin_bf[i] = f2bf(Wlin[i]);
    if (i < 2 * GB * GH)                // h double buffer: 2*128*512
        hbuf2[i] = 0.0f;
    if (i == 0) *ctr = 0u;
}

// ---------------------------------------------------------------------------
// Kernel 1: IG[t][b][3H] = x[b][t][:] @ Wih^T + b   (M=65536, N=1536, K=512)
// One block = one M-tile (16 rows, staged f32->bf16 in LDS) x 8 N-tiles (one
// per wave). grid = 4096 * 12 blocks.
// ---------------------------------------------------------------------------
__global__ void __launch_bounds__(256) gru_ig_gemm_kernel(
    const float* __restrict__ x, const unsigned short* __restrict__ wih_bf,
    const float* __restrict__ bias, float* __restrict__ ig) {
    __shared__ unsigned short abf[16 * GD];   // 16 KB

    int mt = blockIdx.x / 12;                 // 0..4095
    int ng = blockIdx.x % 12;                 // n-group of 8 tiles
    int lane = threadIdx.x & 31;
    int wave = threadIdx.x >> 5;
    int lanerow = lane & 15;
    int asel = (lane & 16) ? 16 : 0;
    int bsel = (lane & 16) ? 16 : 0;

    // stage x tile (16 rows x 512) into LDS as bf16
    const float* xs = x + (size_t)mt * 16 * GD;
    for (int i = threadIdx.x; i < 16 * GD; i += 256)
        abf[i] = f2bf(xs[i]);
    __syncthreads();

    int nt = ng * 8 + wave;                   // 0..95
    int n0 = nt * 16;
    const unsigned short* bp = wih_bf + (size_t)(n0 + lanerow) * GD;

    v8f acc = {0.f, 0.f, 0.f, 0.f, 0.f, 0.f, 0.f, 0.f};
    for (int k0 = 0; k0 < GD; k0 += 32) {
        BfFrag ua, ub;
        const unsigned short* ap = abf + lanerow * GD + k0 + asel;
        ua.q[0] = *(const uint4*)ap;
        ua.q[1] = *(const uint4*)(ap + 8);
        const unsigned short* bq = bp + k0 + bsel;
        ub.q[0] = *(const uint4*)bq;
        ub.q[1] = *(const uint4*)(bq + 8);
        acc = wmma_bf16(ua.v, ub.v, acc);
    }

    int col = n0 + lanerow;
    float bv = bias[col];
    int rbase = mt * 16 + ((lane & 16) ? 8 : 0);
#pragma unroll
    for (int j = 0; j < 8; ++j) {
        int r = rbase + j;                    // flat (b,t) row: r = b*T + t
        int bb = r >> 9;                      // / 512
        int tt = r & 511;
        ig[((size_t)tt * GB + bb) * G3H + col] = acc[j] + bv;
    }
}

// ---------------------------------------------------------------------------
// Kernel 2: persistent GRU recurrence. 16 workgroups = 4 row-groups (32 batch
// rows) x 4 col-groups (128 h-cols). Per wave: one 16-col slice, 2 M-tiles x
// 3 gate accumulators, fused sigmoid/tanh update, double-buffered h, global
// atomic grid barrier per timestep.
// ---------------------------------------------------------------------------
__global__ void __launch_bounds__(256) gru_recurrent_kernel(
    const float* __restrict__ ig, const unsigned short* __restrict__ whh_bf,
    const float* __restrict__ bn, float* __restrict__ hbuf,
    unsigned* __restrict__ ctr) {
    __shared__ unsigned short hbf[32 * GH];   // 32 KB: this WG's 32 h-rows, bf16

    int wg   = blockIdx.x;
    int wgr  = wg >> 2;                       // row group: 32 batch rows
    int wgc  = wg & 3;                        // col group: 128 h columns
    int lane = threadIdx.x & 31;
    int wave = threadIdx.x >> 5;
    int lanerow = lane & 15;
    int asel = (lane & 16) ? 16 : 0;
    int bsel = (lane & 16) ? 16 : 0;

    int hc = wgc * 128 + wave * 16 + lanerow; // this lane's h column
    float bnv = bn[hc];

    const unsigned short* wb0 = whh_bf + (size_t)(0 * GH + hc) * GH;  // r rows
    const unsigned short* wb1 = whh_bf + (size_t)(1 * GH + hc) * GH;  // z rows
    const unsigned short* wb2 = whh_bf + (size_t)(2 * GH + hc) * GH;  // n rows

    for (int t = 0; t < GT; ++t) {
        const float* hr = hbuf + (size_t)(t & 1) * (GB * GH);
        float*       hw = hbuf + (size_t)((t + 1) & 1) * (GB * GH);

        // stage this WG's 32 h-rows (all 512 cols) into LDS as bf16
        const float* hs = hr + (size_t)wgr * 32 * GH;
        for (int i = threadIdx.x; i < 32 * GH; i += 256)
            hbf[i] = f2bf(hs[i]);
        __syncthreads();

        v8f acc00 = {0.f,0.f,0.f,0.f,0.f,0.f,0.f,0.f};
        v8f acc01 = acc00, acc02 = acc00;
        v8f acc10 = acc00, acc11 = acc00, acc12 = acc00;

        for (int k0 = 0; k0 < GH; k0 += 32) {
            BfFrag ua0, ua1, ub;
            const unsigned short* ap0 = hbf + lanerow * GH + k0 + asel;
            const unsigned short* ap1 = ap0 + 16 * GH;
            ua0.q[0] = *(const uint4*)ap0;
            ua0.q[1] = *(const uint4*)(ap0 + 8);
            ua1.q[0] = *(const uint4*)ap1;
            ua1.q[1] = *(const uint4*)(ap1 + 8);

            const unsigned short* bq0 = wb0 + k0 + bsel;
            ub.q[0] = *(const uint4*)bq0;
            ub.q[1] = *(const uint4*)(bq0 + 8);
            acc00 = wmma_bf16(ua0.v, ub.v, acc00);
            acc10 = wmma_bf16(ua1.v, ub.v, acc10);

            const unsigned short* bq1 = wb1 + k0 + bsel;
            ub.q[0] = *(const uint4*)bq1;
            ub.q[1] = *(const uint4*)(bq1 + 8);
            acc01 = wmma_bf16(ua0.v, ub.v, acc01);
            acc11 = wmma_bf16(ua1.v, ub.v, acc11);

            const unsigned short* bq2 = wb2 + k0 + bsel;
            ub.q[0] = *(const uint4*)bq2;
            ub.q[1] = *(const uint4*)(bq2 + 8);
            acc02 = wmma_bf16(ua0.v, ub.v, acc02);
            acc12 = wmma_bf16(ua1.v, ub.v, acc12);
        }

        // fused GRU pointwise update on accumulator registers
        const float* igb = ig + (size_t)t * GB * G3H;
#pragma unroll
        for (int mt = 0; mt < 2; ++mt) {
            int rlb = mt * 16 + ((lane & 16) ? 8 : 0);
#pragma unroll
            for (int j = 0; j < 8; ++j) {
                int rl = rlb + j;                 // local batch row 0..31
                int rg = wgr * 32 + rl;           // global batch row
                const float* igrow = igb + (size_t)rg * G3H + hc;
                float hgr = mt ? acc10[j] : acc00[j];
                float hgz = mt ? acc11[j] : acc01[j];
                float hgn = mt ? acc12[j] : acc02[j];
                float r  = sigmoidf_fast(igrow[0]        + hgr);
                float z  = sigmoidf_fast(igrow[GH]       + hgz);
                float nn = tanhf(igrow[2 * GH] + r * (hgn + bnv));
                float ho = hr[(size_t)rg * GH + hc];
                hw[(size_t)rg * GH + hc] = (1.0f - z) * nn + z * ho;
            }
        }

        // grid-wide barrier (device scope) before next timestep
        __threadfence();
        __syncthreads();
        if (threadIdx.x == 0) {
            __hip_atomic_fetch_add(ctr, 1u, __ATOMIC_ACQ_REL,
                                   __HIP_MEMORY_SCOPE_AGENT);
            unsigned target = (unsigned)(t + 1) * 16u;
            while (__hip_atomic_load(ctr, __ATOMIC_ACQUIRE,
                                     __HIP_MEMORY_SCOPE_AGENT) < target)
                __builtin_amdgcn_s_sleep(8);
        }
        __syncthreads();
    }
}

// ---------------------------------------------------------------------------
// Kernel 3: out = h_final @ Wlin^T + blin, split into (mu | logvar) flat.
// M=128, N=512, K=512 -> 256 16x16 tiles, one per wave, 32 blocks.
// ---------------------------------------------------------------------------
__global__ void __launch_bounds__(256) gru_out_kernel(
    const float* __restrict__ h, const unsigned short* __restrict__ wlin_bf,
    const float* __restrict__ blin, float* __restrict__ out) {
    int lane = threadIdx.x & 31;
    int wave = threadIdx.x >> 5;
    int tid  = blockIdx.x * 8 + wave;         // 0..255
    int mt   = tid >> 5;                      // 0..7
    int nt   = tid & 31;                      // 0..31
    int lanerow = lane & 15;
    int asel = (lane & 16) ? 16 : 0;
    int bsel = (lane & 16) ? 16 : 0;

    const float* ap_base = h + (size_t)(mt * 16 + lanerow) * GH;
    const unsigned short* bp = wlin_bf + (size_t)(nt * 16 + lanerow) * GH;

    v8f acc = {0.f, 0.f, 0.f, 0.f, 0.f, 0.f, 0.f, 0.f};
    for (int k0 = 0; k0 < GH; k0 += 32) {
        BfFrag ua, ub;
        const float* ap = ap_base + k0 + asel;
#pragma unroll
        for (int i = 0; i < 8; ++i) {
            ua.s[i]     = f2bf(ap[i]);
            ua.s[8 + i] = f2bf(ap[8 + i]);
        }
        const unsigned short* bq = bp + k0 + bsel;
        ub.q[0] = *(const uint4*)bq;
        ub.q[1] = *(const uint4*)(bq + 8);
        acc = wmma_bf16(ua.v, ub.v, acc);
    }

    int col = nt * 16 + lanerow;              // 0..511 across (mu|logvar)
    float bv = blin[col];
    int rbase = mt * 16 + ((lane & 16) ? 8 : 0);
#pragma unroll
    for (int j = 0; j < 8; ++j) {
        int row = rbase + j;
        float v = acc[j] + bv;
        if (col < 256) out[row * 256 + col] = v;                    // mu
        else           out[GB * 256 + row * 256 + (col - 256)] = v; // logvar
    }
}

// ---------------------------------------------------------------------------
extern "C" void kernel_launch(void* const* d_in, const int* in_sizes, int n_in,
                              void* d_out, int out_size, void* d_ws, size_t ws_size,
                              hipStream_t stream) {
    const float* x    = (const float*)d_in[0];   // (B, T, D)
    const float* Wih  = (const float*)d_in[1];   // (3H, D)
    const float* Whh  = (const float*)d_in[2];   // (3H, H)
    const float* b    = (const float*)d_in[3];   // (3H)
    const float* bn   = (const float*)d_in[4];   // (H)
    const float* Wlin = (const float*)d_in[5];   // (2L, H)
    const float* blin = (const float*)d_in[6];   // (2L)
    float* out = (float*)d_out;

    char* ws = (char*)d_ws;
    unsigned* ctr            = (unsigned*)ws;                      // 4 B (+pad)
    unsigned short* wih_bf   = (unsigned short*)(ws + 256);        // 1.5 MB
    unsigned short* whh_bf   = wih_bf + (size_t)G3H * GD;          // 1.5 MB
    unsigned short* wlin_bf  = whh_bf + (size_t)G3H * GH;          // 0.5 MB
    float* hbuf              = (float*)(wlin_bf + (size_t)512 * GH); // 2x256 KB
    float* ig                = hbuf + (size_t)2 * GB * GH;         // 402 MB fp32

    gru_prep_kernel<<<(G3H * GD + 255) / 256, 256, 0, stream>>>(
        Wih, Whh, Wlin, wih_bf, whh_bf, wlin_bf, hbuf, ctr);

    gru_ig_gemm_kernel<<<4096 * 12, 256, 0, stream>>>(x, wih_bf, b, ig);

    gru_recurrent_kernel<<<16, 256, 0, stream>>>(ig, whh_bf, bn, hbuf, ctr);

    // after t=511 the final h lives in buffer ((511+1)&1)==0 -> hbuf base
    gru_out_kernel<<<32, 256, 0, stream>>>(hbuf, wlin_bf, blin, out);
}
